// TransformerBlock_41205916237928
// MI455X (gfx1250) — matrix operand order repaired and checked by reading, not compiled, not run on hardware
//
#include <hip/hip_runtime.h>
#include <hip/hip_bf16.h>
#include <math.h>

// Problem constants (match reference)
#define BB    2
#define TT    2048
#define DD    1024
#define HH    16
#define DKV   64
#define DFFN  4096
#define MTOK  (BB * TT)   // 4096 token rows

typedef __attribute__((ext_vector_type(16))) __bf16 v16bf;
typedef __attribute__((ext_vector_type(8)))  float  v8f;

// ---------------------------------------------------------------------------
// WMMA fragment loaders (wave32, 16x16x32 bf16)
// A layout (16-bit A 16x32): lane l holds row M=l%16; elems 0..7 = K=(l/16)*16+{0..7} (order used here, same as B),
// elems 8..15 = K=(l/16)*16+8+{0..7}  -> two contiguous 16B chunks.
// B layout (32x16): lane l holds col N=l%16; elems j = K=(l/16)*16+j -> one 32B chunk.
// C/D layout: lane l, vgpr r -> M = r + 8*(l/16), N = l%16.
// ---------------------------------------------------------------------------
__device__ __forceinline__ v16bf frag_contig(const __bf16* p) {
  union { v16bf v; uint4 u[2]; } f;
  f.u[0] = *(const uint4*)(p);
  f.u[1] = *(const uint4*)(p + 8);
  return f.v;
}
__device__ __forceinline__ v16bf frag_a(const __bf16* p, int hi) {
  union { v16bf v; uint4 u[2]; } f;
  f.u[0] = *(const uint4*)(p + hi * 16);
  f.u[1] = *(const uint4*)(p + hi * 16 + 8);
  return f.v;
}
__device__ __forceinline__ v8f wmma_bf16(v16bf a, v16bf b, v8f c) {
  return __builtin_amdgcn_wmma_f32_16x16x32_bf16(false, a, false, b, (short)0, c,
                                                 false, false);
}

// ---------------------------------------------------------------------------
// DPP16 reductions within each 16-lane row (VALU only, no LDS traffic).
// ---------------------------------------------------------------------------
__device__ __forceinline__ float dpp_max16(float x) {
  int t;
  t = __builtin_amdgcn_update_dpp(__float_as_int(x), __float_as_int(x),
                                  0xB1, 0xf, 0xf, true);
  x = fmaxf(x, __int_as_float(t));
  t = __builtin_amdgcn_update_dpp(__float_as_int(x), __float_as_int(x),
                                  0x4E, 0xf, 0xf, true);
  x = fmaxf(x, __int_as_float(t));
  t = __builtin_amdgcn_update_dpp(__float_as_int(x), __float_as_int(x),
                                  0x141, 0xf, 0xf, true);
  x = fmaxf(x, __int_as_float(t));
  t = __builtin_amdgcn_update_dpp(__float_as_int(x), __float_as_int(x),
                                  0x140, 0xf, 0xf, true);
  x = fmaxf(x, __int_as_float(t));
  return x;
}
__device__ __forceinline__ float dpp_sum16(float x) {
  int t;
  t = __builtin_amdgcn_update_dpp(__float_as_int(x), __float_as_int(x),
                                  0xB1, 0xf, 0xf, true);
  x += __int_as_float(t);
  t = __builtin_amdgcn_update_dpp(__float_as_int(x), __float_as_int(x),
                                  0x4E, 0xf, 0xf, true);
  x += __int_as_float(t);
  t = __builtin_amdgcn_update_dpp(__float_as_int(x), __float_as_int(x),
                                  0x141, 0xf, 0xf, true);
  x += __int_as_float(t);
  t = __builtin_amdgcn_update_dpp(__float_as_int(x), __float_as_int(x),
                                  0x140, 0xf, 0xf, true);
  x += __int_as_float(t);
  return x;
}

// ---------------------------------------------------------------------------
// Async global -> LDS copy (gfx1250 ASYNCcnt path).
// lds_off: workgroup-relative LDS byte address (low 32 bits of generic ptr).
// ---------------------------------------------------------------------------
__device__ __forceinline__ void async_copy_b128(const void* gsrc, void* lds_dst) {
  const unsigned dst = (unsigned)(uintptr_t)lds_dst;
  const unsigned long long src = (unsigned long long)(uintptr_t)gsrc;
  asm volatile("global_load_async_to_lds_b128 %0, %1, off"
               :: "v"(dst), "v"(src) : "memory");
}
__device__ __forceinline__ void async_wait0() {
  asm volatile("s_wait_asynccnt 0x0" ::: "memory");
}

// ---------------------------------------------------------------------------
// Weight convert + transpose: W[K][N] fp32 (row major) -> WT[N][K] bf16
// ---------------------------------------------------------------------------
__global__ __launch_bounds__(256)
void transpose_bf16(const float* __restrict__ w, __bf16* __restrict__ wT,
                    int K, int N) {
  __shared__ float tile[32][33];
  const int k0 = blockIdx.y * 32, n0 = blockIdx.x * 32;
  const int tx = threadIdx.x & 31, ty = threadIdx.x >> 5;  // 32 x 8
#pragma unroll
  for (int r = ty; r < 32; r += 8)
    tile[r][tx] = w[(size_t)(k0 + r) * N + n0 + tx];
  __syncthreads();
#pragma unroll
  for (int r = ty; r < 32; r += 8)
    wT[(size_t)(n0 + r) * K + k0 + tx] = (__bf16)tile[tx][r];
}

// ---------------------------------------------------------------------------
// LayerNorm (unbiased std, eps added to std) -> bf16 output
// ---------------------------------------------------------------------------
__global__ __launch_bounds__(256)
void layernorm_bf16(const float* __restrict__ x, const float* __restrict__ g,
                    const float* __restrict__ b, __bf16* __restrict__ out,
                    int D) {
  const int row = blockIdx.x;
  const float* xr = x + (size_t)row * D;
  float s = 0.f, sq = 0.f;
  for (int i = threadIdx.x; i < D; i += 256) {
    float v = xr[i];
    s += v;
    sq += v * v;
  }
#pragma unroll
  for (int off = 16; off > 0; off >>= 1) {
    s += __shfl_down(s, off, 32);
    sq += __shfl_down(sq, off, 32);
  }
  __shared__ float r0[8], r1[8];
  const int lane = threadIdx.x & 31, w = threadIdx.x >> 5;
  if (lane == 0) { r0[w] = s; r1[w] = sq; }
  __syncthreads();
  if (threadIdx.x == 0) {
    float ts = 0.f, tq = 0.f;
#pragma unroll
    for (int i = 0; i < 8; i++) { ts += r0[i]; tq += r1[i]; }
    r0[0] = ts; r1[0] = tq;
  }
  __syncthreads();
  s = r0[0]; sq = r1[0];
  const float mean = s / (float)D;
  float var = (sq - s * mean) / (float)(D - 1);
  var = fmaxf(var, 0.f);
  const float inv = 1.0f / (sqrtf(var) + 1e-6f);
  for (int i = threadIdx.x; i < D; i += 256) {
    float v = (xr[i] - mean) * g[i] * inv + b[i];
    out[(size_t)row * D + i] = (__bf16)v;
  }
}

// ---------------------------------------------------------------------------
// Generic bf16 WMMA GEMM:  C[M][N] = A[M][K] @ BT[N][K]^T  (+epilogue)
// One wave computes a 32x64 block (2 A-frags reuse 4 B-frags -> 8 WMMA/k-step).
// The 32x32 A tile (shared by all 4 waves) is staged into LDS with async
// global->LDS copies, double-buffered to overlap copy with WMMA compute.
// block = 128 (4 waves), grid(N/256, M/32).
// ---------------------------------------------------------------------------
template <bool BIAS, bool GELU, bool RES, bool OUTF, bool OUTB, bool TRANS>
__global__ __launch_bounds__(128)
void gemm_bf16_wmma(const __bf16* __restrict__ A, const __bf16* __restrict__ Bt,
                    const float* __restrict__ bias, const float* __restrict__ res,
                    float* __restrict__ outf, __bf16* __restrict__ outb,
                    int M, int N, int K) {
  const int lane = threadIdx.x & 31;
  const int wid  = threadIdx.x >> 5;
  const int lo = lane & 15, hi = lane >> 4;
  const int mt = blockIdx.y;                 // 32-row tile
  const int nb = (blockIdx.x * 4 + wid) * 64;

  const __bf16* bp = Bt + (size_t)(nb + lo) * K;

  // A tile staging: 32 rows x 32 k-elems bf16 = 2KB per buffer
  __shared__ __align__(16) __bf16 atile[2][32 * 32];
  const int crow = threadIdx.x >> 2;         // 0..31
  const int cchk = threadIdx.x & 3;          // 0..3 (16B chunks)
  const __bf16* asrc = A + (size_t)(mt * 32 + crow) * K + cchk * 8;

  v8f acc[2][4];
#pragma unroll
  for (int mi = 0; mi < 2; mi++)
#pragma unroll
    for (int t = 0; t < 4; t++)
#pragma unroll
      for (int r = 0; r < 8; r++) acc[mi][t][r] = 0.f;

  // prologue: stage k=0 tile
  async_copy_b128(asrc, &atile[0][crow * 32 + cchk * 8]);
  async_wait0();
  __syncthreads();

  int cur = 0;
  for (int k = 0; k < K; k += 32) {
    if (k + 32 < K) {  // issue next tile copy; overlaps with WMMAs below
      async_copy_b128(asrc + k + 32, &atile[cur ^ 1][crow * 32 + cchk * 8]);
    }
    if (k + 256 < K) {  // stream-ahead prefetch of the weight rows
#pragma unroll
      for (int t = 0; t < 4; t++)
        __builtin_prefetch(bp + (size_t)t * 16 * K + k + 256 + hi * 16, 0, 1);
    }
    const __bf16* at = &atile[cur][0];
    const v16bf a0 = frag_a(at + lo * 32, hi);
    const v16bf a1 = frag_a(at + (16 + lo) * 32, hi);
#pragma unroll
    for (int t = 0; t < 4; t++) {
      const v16bf bf = frag_contig(bp + (size_t)t * 16 * K + k + hi * 16);
      acc[0][t] = wmma_bf16(a0, bf, acc[0][t]);
      acc[1][t] = wmma_bf16(a1, bf, acc[1][t]);
    }
    async_wait0();
    __syncthreads();
    cur ^= 1;
  }

#pragma unroll
  for (int mi = 0; mi < 2; mi++) {
#pragma unroll
    for (int t = 0; t < 4; t++) {
#pragma unroll
      for (int r = 0; r < 8; r++) {
        const int m = mt * 32 + mi * 16 + r + 8 * hi;
        const int n = nb + t * 16 + lo;
        float v = acc[mi][t][r];
        if (BIAS) v += bias[n];
        if (GELU) v = 0.5f * v * (1.0f + erff(v * 0.70710678118654752f));
        if (RES)  v += res[(size_t)m * N + n];
        if (OUTF) outf[(size_t)m * N + n] = v;
        if (OUTB) {
          if (TRANS) outb[(size_t)n * M + m] = (__bf16)v;
          else       outb[(size_t)m * N + n] = (__bf16)v;
        }
      }
    }
  }
}

// ---------------------------------------------------------------------------
// Flash attention (causal, scale 1/8). One wave per 16-query tile per (b,h).
// Q,K bf16 [MTOK][H*64] row-major; VT bf16 [H*64][MTOK]; O bf16 [MTOK][H*64].
// block = 128 (4 waves), grid = B*H*(T/16)/4.
// ---------------------------------------------------------------------------
__global__ __launch_bounds__(128)
void attn_flash(const __bf16* __restrict__ Q, const __bf16* __restrict__ Kb,
                const __bf16* __restrict__ VT, __bf16* __restrict__ O) {
  const int HD = HH * DKV;  // 1024
  const int lane = threadIdx.x & 31;
  const int wid  = threadIdx.x >> 5;
  const int lo = lane & 15, hi = lane >> 4;

  const int gt = blockIdx.x * 4 + wid;          // global query tile id
  const int tilesPerSeq = TT / 16;              // 128
  const int b  = gt / (HH * tilesPerSeq);
  const int rm = gt % (HH * tilesPerSeq);
  const int h  = rm / tilesPerSeq;
  const int qi = rm % tilesPerSeq;

  // Q fragments (A layout) for k-steps 0 and 1 over d=64
  const __bf16* qp = Q + (size_t)(b * TT + qi * 16 + lo) * HD + h * DKV;
  const v16bf qa0 = frag_a(qp, hi);
  const v16bf qa1 = frag_a(qp + 32, hi);

  v8f acc[4];
#pragma unroll
  for (int t = 0; t < 4; t++)
#pragma unroll
    for (int r = 0; r < 8; r++) acc[t][r] = 0.f;
  float mrun[8], lrun[8];
#pragma unroll
  for (int r = 0; r < 8; r++) { mrun[r] = -INFINITY; lrun[r] = 0.f; }

  __shared__ __bf16 plds[4][16 * 32];
  __bf16* pl = plds[wid];

  const int jend = (qi * 16 + 15) >> 5;
  for (int j = 0; j <= jend; j++) {
    const int kc = j * 32;
    const __bf16* kp0 = Kb + (size_t)(b * TT + kc + lo) * HD + h * DKV;
    const __bf16* kp1 = kp0 + (size_t)16 * HD;

    v8f s0, s1;
#pragma unroll
    for (int r = 0; r < 8; r++) { s0[r] = 0.f; s1[r] = 0.f; }
    s0 = wmma_bf16(qa0, frag_contig(kp0 + hi * 16), s0);
    s0 = wmma_bf16(qa1, frag_contig(kp0 + 32 + hi * 16), s0);
    s1 = wmma_bf16(qa0, frag_contig(kp1 + hi * 16), s1);
    s1 = wmma_bf16(qa1, frag_contig(kp1 + 32 + hi * 16), s1);

    float f0[8], f1[8];
#pragma unroll
    for (int r = 0; r < 8; r++) {
      const int qpos = qi * 16 + r + 8 * hi;
      f0[r] = (kc + lo      <= qpos) ? s0[r] * 0.125f : -INFINITY;
      f1[r] = (kc + 16 + lo <= qpos) ? s1[r] * 0.125f : -INFINITY;
    }

#pragma unroll
    for (int r = 0; r < 8; r++) {
      const float mx    = dpp_max16(fmaxf(f0[r], f1[r]));
      const float mnew  = fmaxf(mrun[r], mx);
      const float alpha = __expf(mrun[r] - mnew);
      mrun[r] = mnew;
      const float p0 = __expf(f0[r] - mnew);
      const float p1 = __expf(f1[r] - mnew);
      const float rs = dpp_sum16(p0 + p1);
      lrun[r] = lrun[r] * alpha + rs;
#pragma unroll
      for (int t = 0; t < 4; t++) acc[t][r] *= alpha;
      // store P tile (C layout -> row-major LDS) as bf16
      pl[(r + 8 * hi) * 32 + lo]      = (__bf16)p0;
      pl[(r + 8 * hi) * 32 + 16 + lo] = (__bf16)p1;
    }
    asm volatile("s_wait_dscnt 0" ::: "memory");

    // reload P in A layout (16x32)
    const v16bf pa = frag_a(pl + lo * 32, hi);

#pragma unroll
    for (int t = 0; t < 4; t++) {
      const __bf16* vp = VT + (size_t)(h * DKV + t * 16 + lo) * MTOK +
                         b * TT + kc + hi * 16;
      acc[t] = wmma_bf16(pa, frag_contig(vp), acc[t]);
    }
    asm volatile("s_wait_dscnt 0" ::: "memory");
  }

  float inv[8];
#pragma unroll
  for (int r = 0; r < 8; r++) inv[r] = 1.0f / lrun[r];
#pragma unroll
  for (int t = 0; t < 4; t++) {
#pragma unroll
    for (int r = 0; r < 8; r++) {
      const size_t m = (size_t)(b * TT + qi * 16 + r + 8 * hi);
      O[m * HD + h * DKV + t * 16 + lo] = (__bf16)(acc[t][r] * inv[r]);
    }
  }
}

// ---------------------------------------------------------------------------
// Launch
// ---------------------------------------------------------------------------
extern "C" void kernel_launch(void* const* d_in, const int* in_sizes, int n_in,
                              void* d_out, int out_size, void* d_ws,
                              size_t ws_size, hipStream_t stream) {
  (void)in_sizes; (void)n_in; (void)out_size; (void)ws_size;
  const float* x     = (const float*)d_in[0];
  const float* w_q   = (const float*)d_in[1];
  const float* w_k   = (const float*)d_in[2];
  const float* w_v   = (const float*)d_in[3];
  const float* w_o   = (const float*)d_in[4];
  const float* b_o   = (const float*)d_in[5];
  const float* w_fc1 = (const float*)d_in[6];
  const float* b_fc1 = (const float*)d_in[7];
  const float* w_fc2 = (const float*)d_in[8];
  const float* b_fc2 = (const float*)d_in[9];
  const float* ln1_g = (const float*)d_in[10];
  const float* ln1_b = (const float*)d_in[11];
  const float* ln2_g = (const float*)d_in[12];
  const float* ln2_b = (const float*)d_in[13];
  float* out = (float*)d_out;

  char* ws = (char*)d_ws;
  auto carve = [&](size_t bytes) -> void* {
    void* p = (void*)ws;
    ws += (bytes + 255) & ~(size_t)255;
    return p;
  };
  __bf16* wqT   = (__bf16*)carve((size_t)DD * DD * 2);
  __bf16* wkT   = (__bf16*)carve((size_t)DD * DD * 2);
  __bf16* wvT   = (__bf16*)carve((size_t)DD * DD * 2);
  __bf16* woT   = (__bf16*)carve((size_t)DD * DD * 2);
  __bf16* wfc1T = (__bf16*)carve((size_t)DD * DFFN * 2);
  __bf16* wfc2T = (__bf16*)carve((size_t)DFFN * DD * 2);
  __bf16* hbf   = (__bf16*)carve((size_t)MTOK * DD * 2);
  __bf16* qb    = (__bf16*)carve((size_t)MTOK * DD * 2);
  __bf16* kb    = (__bf16*)carve((size_t)MTOK * DD * 2);
  __bf16* vTb   = (__bf16*)carve((size_t)DD * MTOK * 2);
  __bf16* ctxb  = (__bf16*)carve((size_t)MTOK * DD * 2);
  float*  x1    = (float*)carve((size_t)MTOK * DD * 4);
  __bf16* h2bf  = (__bf16*)carve((size_t)MTOK * DD * 2);
  __bf16* actb  = (__bf16*)carve((size_t)MTOK * DFFN * 2);

  const dim3 tb(256);
  // Weight convert/transpose (each call; deterministic)
  transpose_bf16<<<dim3(DD / 32, DD / 32), tb, 0, stream>>>(w_q, wqT, DD, DD);
  transpose_bf16<<<dim3(DD / 32, DD / 32), tb, 0, stream>>>(w_k, wkT, DD, DD);
  transpose_bf16<<<dim3(DD / 32, DD / 32), tb, 0, stream>>>(w_v, wvT, DD, DD);
  transpose_bf16<<<dim3(DD / 32, DD / 32), tb, 0, stream>>>(w_o, woT, DD, DD);
  transpose_bf16<<<dim3(DFFN / 32, DD / 32), tb, 0, stream>>>(w_fc1, wfc1T, DD, DFFN);
  transpose_bf16<<<dim3(DD / 32, DFFN / 32), tb, 0, stream>>>(w_fc2, wfc2T, DFFN, DD);

  // LN1
  layernorm_bf16<<<dim3(MTOK), tb, 0, stream>>>(x, ln1_g, ln1_b, hbf, DD);

  const dim3 gb(128);
  // Q, K, V projections (V stored transposed per column: [H*64][MTOK])
  gemm_bf16_wmma<false, false, false, false, true, false>
      <<<dim3(DD / 256, MTOK / 32), gb, 0, stream>>>(hbf, wqT, nullptr, nullptr,
                                                     nullptr, qb, MTOK, DD, DD);
  gemm_bf16_wmma<false, false, false, false, true, false>
      <<<dim3(DD / 256, MTOK / 32), gb, 0, stream>>>(hbf, wkT, nullptr, nullptr,
                                                     nullptr, kb, MTOK, DD, DD);
  gemm_bf16_wmma<false, false, false, false, true, true>
      <<<dim3(DD / 256, MTOK / 32), gb, 0, stream>>>(hbf, wvT, nullptr, nullptr,
                                                     nullptr, vTb, MTOK, DD, DD);

  // Causal flash attention
  attn_flash<<<dim3(BB * HH * (TT / 16) / 4), gb, 0, stream>>>(qb, kb, vTb, ctxb);

  // Output projection + bias + residual(x) -> x1 (fp32)
  gemm_bf16_wmma<true, false, true, true, false, false>
      <<<dim3(DD / 256, MTOK / 32), gb, 0, stream>>>(ctxb, woT, b_o, x, x1,
                                                     nullptr, MTOK, DD, DD);
  // LN2
  layernorm_bf16<<<dim3(MTOK), tb, 0, stream>>>(x1, ln2_g, ln2_b, h2bf, DD);

  // FC1 + bias + exact GELU -> bf16 activations
  gemm_bf16_wmma<true, true, false, false, true, false>
      <<<dim3(DFFN / 256, MTOK / 32), gb, 0, stream>>>(h2bf, wfc1T, b_fc1,
                                                       nullptr, nullptr, actb,
                                                       MTOK, DFFN, DD);
  // FC2 + bias + residual(x1) -> out (fp32)
  gemm_bf16_wmma<true, false, true, true, false, false>
      <<<dim3(DD / 256, MTOK / 32), gb, 0, stream>>>(actb, wfc2T, b_fc2, x1, out,
                                                     nullptr, MTOK, DD, DFFN);
}
